// TorchMoe_64089501991105
// MI455X (gfx1250) — compile-verified, matrix operand order checked by reading, not run on hardware
//
#include <hip/hip_runtime.h>
#include <hip/hip_bf16.h>

// Problem constants (match reference)
#define G_  4
#define S_  1024
#define D_  1024
#define E_  16
#define K_  2
#define NT  (G_ * S_)      // 4096 tokens
#define CAP (NT * K_)      // 8192 routed slots total

// GEMM tiling: 256 threads = 8 wave32s in a 4x2 grid, each wave owns 32x32 out
#define BM 128
#define BN 64
#define BK 32
#define THREADS 256
#define LDP (BK + 4)       // LDS pitch (bf16): 36*2B = 72B -> 8B-aligned rows, spread banks

typedef __attribute__((ext_vector_type(16))) __bf16 v16bf;
typedef __attribute__((ext_vector_type(8)))  float  v8f;
typedef __attribute__((ext_vector_type(4)))  float  f4;
typedef __attribute__((ext_vector_type(2)))  float  f2v;
typedef __attribute__((ext_vector_type(4)))  __bf16 bf4;
typedef __attribute__((ext_vector_type(2)))  __bf16 bf2v;

#define WAIT_ASYNC() asm volatile("s_wait_asynccnt 0x0" ::: "memory")

// per-lane async copy of 8 bytes global -> LDS, tracked on ASYNCcnt
__device__ __forceinline__ void async_copy_b64(unsigned lds_off, const void* gptr) {
  unsigned long long ga = (unsigned long long)(uintptr_t)gptr;
  asm volatile("global_load_async_to_lds_b64 %0, %1, off"
               :: "v"(lds_off), "v"(ga) : "memory");
}

// ---- WMMA fragment index helpers (CDNA5 ISA §7.12.2, 16-bit 16x32 A layout) ----
__device__ __forceinline__ int frag_k(int lane, int i) {
  int base = (lane & 16) ? 8 : 0;
  return base + ((i < 8) ? i : (i + 8));
}

__device__ __forceinline__ v16bf load_frag(const __bf16* rowptr, int lane) {
  v16bf v;
#pragma unroll
  for (int i = 0; i < 16; ++i) v[i] = rowptr[frag_k(lane, i)];
  return v;
}

__device__ __forceinline__ v8f wmma_bf16(v16bf a, v16bf b, v8f c) {
  return __builtin_amdgcn_wmma_f32_16x16x32_bf16(
      false, a, false, b, (short)0, c, false, false);
}

__device__ __forceinline__ float silu_fast(float v) {
  return v * __builtin_amdgcn_rcpf(1.0f + __expf(-v));
}

// vector fptrunc -> packed bf16 converts (v_cvt_pk_bf16_f32)
__device__ __forceinline__ void store_bf4_cvt(void* dst, f4 v) {
  *(bf4*)dst = __builtin_convertvector(v, bf4);
}
__device__ __forceinline__ void store_bf2_cvt(void* dst, float lo, float hi) {
  f2v p = {lo, hi};
  *(bf2v*)dst = __builtin_convertvector(p, bf2v);
}

// =====================  Routing kernels  =====================

__global__ void k_zero(int* counts, int* fill) {
  int t = threadIdx.x;
  if (t < E_) { counts[t] = 0; fill[t] = 0; }
}

__global__ void k_count(const int* __restrict__ idxs, int* __restrict__ counts) {
  int i = blockIdx.x * blockDim.x + threadIdx.x;
  atomicAdd(&counts[idxs[i] & (E_ - 1)], 1);
}

__global__ void k_offsets(const int* __restrict__ counts, int* __restrict__ offs) {
  if (threadIdx.x == 0) {
    int s = 0;
    for (int e = 0; e < E_; ++e) { offs[e] = s; s += counts[e]; }
  }
}

__global__ void k_fill(const int* __restrict__ idxs, const float* __restrict__ w,
                       const int* __restrict__ offs, int* __restrict__ fill,
                       int* __restrict__ tokens, float* __restrict__ wts) {
  int i = blockIdx.x * blockDim.x + threadIdx.x;
  int e = idxs[i] & (E_ - 1);
  int slot = atomicAdd(&fill[e], 1);
  int p = offs[e] + slot;
  tokens[p] = i / K_;
  wts[p] = w[i];
}

// =====================  Shared expert: fused gate+up  =====================

__global__ __launch_bounds__(THREADS)
void k_gateup_shared(const float* __restrict__ x, const float* __restrict__ sg,
                     const float* __restrict__ su, __bf16* __restrict__ hs) {
  __shared__ __bf16 As[2][BM][LDP];
  __shared__ __bf16 Bg[2][BN][LDP];
  __shared__ __bf16 Bu[2][BN][LDP];
  const int tid = threadIdx.x;
  const int lane = tid & 31, wid = tid >> 5;
  const int wm = wid & 3, wn = wid >> 2;
  const int m0 = blockIdx.y * BM;
  const int n0 = blockIdx.x * BN;

  const float* aptr[4]; int ar[4], ac[4];
#pragma unroll
  for (int j = 0; j < 4; ++j) {
    int v = tid + j * THREADS;
    ar[j] = v >> 3; ac[j] = (v & 7) * 4;
    aptr[j] = x + (size_t)(m0 + ar[j]) * D_ + ac[j];
  }
  const int bk2 = (tid >> 4) * 2;
  const int bnc = (tid & 15) * 4;
  const float* bgp = sg + (size_t)bk2 * D_ + n0 + bnc;
  const float* bup = su + (size_t)bk2 * D_ + n0 + bnc;

  f4 ra[4], rg0, rg1, ru0, ru1;
  auto gload = [&](int k0) {
#pragma unroll
    for (int j = 0; j < 4; ++j) ra[j] = *(const f4*)(aptr[j] + k0);
    rg0 = *(const f4*)(bgp + (size_t)k0 * D_);
    rg1 = *(const f4*)(bgp + (size_t)(k0 + 1) * D_);
    ru0 = *(const f4*)(bup + (size_t)k0 * D_);
    ru1 = *(const f4*)(bup + (size_t)(k0 + 1) * D_);
  };
  auto lstore = [&](int pb) {
#pragma unroll
    for (int j = 0; j < 4; ++j) store_bf4_cvt(&As[pb][ar[j]][ac[j]], ra[j]);
#pragma unroll
    for (int t = 0; t < 4; ++t) {
      store_bf2_cvt(&Bg[pb][bnc + t][bk2], rg0[t], rg1[t]);
      store_bf2_cvt(&Bu[pb][bnc + t][bk2], ru0[t], ru1[t]);
    }
  };

  v8f accg[2][2] = {}; v8f accu[2][2] = {};
  gload(0);
  lstore(0);
  __syncthreads();
  int pb = 0;
  for (int k0 = 0; k0 < D_; k0 += BK) {
    const bool more = (k0 + BK) < D_;
    if (more) gload(k0 + BK);
    v16bf af[2], bgf[2], buf[2];
#pragma unroll
    for (int mi = 0; mi < 2; ++mi) af[mi] = load_frag(&As[pb][wm * 32 + mi * 16 + (lane & 15)][0], lane);
#pragma unroll
    for (int ni = 0; ni < 2; ++ni) {
      bgf[ni] = load_frag(&Bg[pb][wn * 32 + ni * 16 + (lane & 15)][0], lane);
      buf[ni] = load_frag(&Bu[pb][wn * 32 + ni * 16 + (lane & 15)][0], lane);
    }
#pragma unroll
    for (int mi = 0; mi < 2; ++mi)
#pragma unroll
      for (int ni = 0; ni < 2; ++ni) {
        accg[mi][ni] = wmma_bf16(af[mi], bgf[ni], accg[mi][ni]);
        accu[mi][ni] = wmma_bf16(af[mi], buf[ni], accu[mi][ni]);
      }
    if (more) lstore(pb ^ 1);
    __syncthreads();
    pb ^= 1;
  }

#pragma unroll
  for (int mi = 0; mi < 2; ++mi)
#pragma unroll
    for (int ni = 0; ni < 2; ++ni)
#pragma unroll
      for (int i = 0; i < 8; ++i) {
        int row = m0 + wm * 32 + mi * 16 + i + ((lane & 16) ? 8 : 0);
        int col = n0 + wn * 32 + ni * 16 + (lane & 15);
        hs[(size_t)row * D_ + col] = (__bf16)(silu_fast(accg[mi][ni][i]) * accu[mi][ni][i]);
      }
}

// =====================  Shared expert: down projection  =====================
// A tiles (bf16) stream via GLOBAL_LOAD_ASYNC_TO_LDS_B64 on ASYNCcnt

__global__ __launch_bounds__(THREADS)
void k_down_shared(const __bf16* __restrict__ hs, const float* __restrict__ sd,
                   float* __restrict__ out) {
  __shared__ __bf16 As[2][BM][LDP];
  __shared__ __bf16 Bt[2][BN][LDP];
  const int tid = threadIdx.x;
  const int lane = tid & 31, wid = tid >> 5;
  const int wm = wid & 3, wn = wid >> 2;
  const int m0 = blockIdx.y * BM;
  const int n0 = blockIdx.x * BN;

  const __bf16* aptr[4]; unsigned aoff[2][4];
#pragma unroll
  for (int j = 0; j < 4; ++j) {
    int v = tid + j * THREADS;
    int r = v >> 3, c = (v & 7) * 4;
    aptr[j] = hs + (size_t)(m0 + r) * D_ + c;
    aoff[0][j] = (unsigned)(uintptr_t)&As[0][r][c];
    aoff[1][j] = (unsigned)(uintptr_t)&As[1][r][c];
  }
  const int bk2 = (tid >> 4) * 2;
  const int bnc = (tid & 15) * 4;
  const float* bp = sd + (size_t)bk2 * D_ + n0 + bnc;

  f4 rb0, rb1;
  auto asyncA = [&](int b, int k0) {
#pragma unroll
    for (int j = 0; j < 4; ++j) async_copy_b64(aoff[b][j], aptr[j] + k0);
  };
  auto gloadB = [&](int k0) {
    rb0 = *(const f4*)(bp + (size_t)k0 * D_);
    rb1 = *(const f4*)(bp + (size_t)(k0 + 1) * D_);
  };
  auto storeB = [&](int pb) {
#pragma unroll
    for (int t = 0; t < 4; ++t)
      store_bf2_cvt(&Bt[pb][bnc + t][bk2], rb0[t], rb1[t]);
  };

  v8f acc[2][2] = {};
  asyncA(0, 0);
  gloadB(0);
  storeB(0);
  WAIT_ASYNC();
  __syncthreads();
  int pb = 0;
  for (int k0 = 0; k0 < D_; k0 += BK) {
    const bool more = (k0 + BK) < D_;
    if (more) { asyncA(pb ^ 1, k0 + BK); gloadB(k0 + BK); }
    v16bf af[2], bf[2];
#pragma unroll
    for (int mi = 0; mi < 2; ++mi) af[mi] = load_frag(&As[pb][wm * 32 + mi * 16 + (lane & 15)][0], lane);
#pragma unroll
    for (int ni = 0; ni < 2; ++ni) bf[ni] = load_frag(&Bt[pb][wn * 32 + ni * 16 + (lane & 15)][0], lane);
#pragma unroll
    for (int mi = 0; mi < 2; ++mi)
#pragma unroll
      for (int ni = 0; ni < 2; ++ni)
        acc[mi][ni] = wmma_bf16(af[mi], bf[ni], acc[mi][ni]);
    if (more) storeB(pb ^ 1);
    WAIT_ASYNC();
    __syncthreads();
    pb ^= 1;
  }

#pragma unroll
  for (int mi = 0; mi < 2; ++mi)
#pragma unroll
    for (int ni = 0; ni < 2; ++ni)
#pragma unroll
      for (int i = 0; i < 8; ++i) {
        int row = m0 + wm * 32 + mi * 16 + i + ((lane & 16) ? 8 : 0);
        int col = n0 + wn * 32 + ni * 16 + (lane & 15);
        out[(size_t)row * D_ + col] = acc[mi][ni][i];
      }
}

// =====================  Routed experts: fused gate+up (grouped GEMM, gather)  ==========

__global__ __launch_bounds__(THREADS)
void k_gateup_routed(const float* __restrict__ x, const float* __restrict__ Wg,
                     const float* __restrict__ Wu, const int* __restrict__ counts,
                     const int* __restrict__ offs, const int* __restrict__ tokens,
                     __bf16* __restrict__ hr) {
  const int e = blockIdx.z;
  const int cnt = counts[e];
  const int m0 = blockIdx.y * BM;
  if (m0 >= cnt) return;
  const int off = offs[e];
  const float* __restrict__ wg = Wg + (size_t)e * D_ * D_;
  const float* __restrict__ wu = Wu + (size_t)e * D_ * D_;

  __shared__ __bf16 As[2][BM][LDP];
  __shared__ __bf16 Bg[2][BN][LDP];
  __shared__ __bf16 Bu[2][BN][LDP];
  const int tid = threadIdx.x;
  const int lane = tid & 31, wid = tid >> 5;
  const int wm = wid & 3, wn = wid >> 2;
  const int n0 = blockIdx.x * BN;

  const float* aptr[4]; int ar[4], ac[4];
#pragma unroll
  for (int j = 0; j < 4; ++j) {
    int v = tid + j * THREADS;
    ar[j] = v >> 3; ac[j] = (v & 7) * 4;
    int lr = m0 + ar[j];
    aptr[j] = (lr < cnt) ? (x + (size_t)tokens[off + lr] * D_ + ac[j]) : nullptr;
  }
  const int bk2 = (tid >> 4) * 2;
  const int bnc = (tid & 15) * 4;
  const float* bgp = wg + (size_t)bk2 * D_ + n0 + bnc;
  const float* bup = wu + (size_t)bk2 * D_ + n0 + bnc;

  f4 ra[4], rg0, rg1, ru0, ru1;
  const f4 fz = {0.f, 0.f, 0.f, 0.f};
  auto gload = [&](int k0) {
#pragma unroll
    for (int j = 0; j < 4; ++j) ra[j] = aptr[j] ? *(const f4*)(aptr[j] + k0) : fz;
    rg0 = *(const f4*)(bgp + (size_t)k0 * D_);
    rg1 = *(const f4*)(bgp + (size_t)(k0 + 1) * D_);
    ru0 = *(const f4*)(bup + (size_t)k0 * D_);
    ru1 = *(const f4*)(bup + (size_t)(k0 + 1) * D_);
  };
  auto lstore = [&](int pb) {
#pragma unroll
    for (int j = 0; j < 4; ++j) store_bf4_cvt(&As[pb][ar[j]][ac[j]], ra[j]);
#pragma unroll
    for (int t = 0; t < 4; ++t) {
      store_bf2_cvt(&Bg[pb][bnc + t][bk2], rg0[t], rg1[t]);
      store_bf2_cvt(&Bu[pb][bnc + t][bk2], ru0[t], ru1[t]);
    }
  };

  v8f accg[2][2] = {}; v8f accu[2][2] = {};
  gload(0);
  lstore(0);
  __syncthreads();
  int pb = 0;
  for (int k0 = 0; k0 < D_; k0 += BK) {
    const bool more = (k0 + BK) < D_;
    if (more) gload(k0 + BK);
    v16bf af[2], bgf[2], buf[2];
#pragma unroll
    for (int mi = 0; mi < 2; ++mi) af[mi] = load_frag(&As[pb][wm * 32 + mi * 16 + (lane & 15)][0], lane);
#pragma unroll
    for (int ni = 0; ni < 2; ++ni) {
      bgf[ni] = load_frag(&Bg[pb][wn * 32 + ni * 16 + (lane & 15)][0], lane);
      buf[ni] = load_frag(&Bu[pb][wn * 32 + ni * 16 + (lane & 15)][0], lane);
    }
#pragma unroll
    for (int mi = 0; mi < 2; ++mi)
#pragma unroll
      for (int ni = 0; ni < 2; ++ni) {
        accg[mi][ni] = wmma_bf16(af[mi], bgf[ni], accg[mi][ni]);
        accu[mi][ni] = wmma_bf16(af[mi], buf[ni], accu[mi][ni]);
      }
    if (more) lstore(pb ^ 1);
    __syncthreads();
    pb ^= 1;
  }

#pragma unroll
  for (int mi = 0; mi < 2; ++mi)
#pragma unroll
    for (int i = 0; i < 8; ++i) {
      int lr = m0 + wm * 32 + mi * 16 + i + ((lane & 16) ? 8 : 0);
      if (lr < cnt) {
#pragma unroll
        for (int ni = 0; ni < 2; ++ni) {
          int col = n0 + wn * 32 + ni * 16 + (lane & 15);
          hr[(size_t)(off + lr) * D_ + col] =
              (__bf16)(silu_fast(accg[mi][ni][i]) * accu[mi][ni][i]);
        }
      }
    }
}

// =====================  Routed experts: down projection + weighted scatter  ==========
// A tiles (bf16) stream via GLOBAL_LOAD_ASYNC_TO_LDS_B64; masked rows clamp to a
// valid row (their outputs are never stored, GEMM rows are independent).

__global__ __launch_bounds__(THREADS)
void k_down_routed(const __bf16* __restrict__ hr, const float* __restrict__ Wd,
                   const int* __restrict__ counts, const int* __restrict__ offs,
                   const int* __restrict__ tokens, const float* __restrict__ wts,
                   float* __restrict__ out) {
  const int e = blockIdx.z;
  const int cnt = counts[e];
  const int m0 = blockIdx.y * BM;
  if (m0 >= cnt) return;
  const int off = offs[e];
  const float* __restrict__ wd = Wd + (size_t)e * D_ * D_;

  __shared__ __bf16 As[2][BM][LDP];
  __shared__ __bf16 Bt[2][BN][LDP];
  const int tid = threadIdx.x;
  const int lane = tid & 31, wid = tid >> 5;
  const int wm = wid & 3, wn = wid >> 2;
  const int n0 = blockIdx.x * BN;

  const __bf16* aptr[4]; unsigned aoff[2][4];
#pragma unroll
  for (int j = 0; j < 4; ++j) {
    int v = tid + j * THREADS;
    int r = v >> 3, c = (v & 7) * 4;
    int lr = m0 + r;
    int lrc = (lr < cnt) ? lr : (cnt - 1);     // clamp: row never stored if masked
    aptr[j] = hr + (size_t)(off + lrc) * D_ + c;
    aoff[0][j] = (unsigned)(uintptr_t)&As[0][r][c];
    aoff[1][j] = (unsigned)(uintptr_t)&As[1][r][c];
  }
  const int bk2 = (tid >> 4) * 2;
  const int bnc = (tid & 15) * 4;
  const float* bp = wd + (size_t)bk2 * D_ + n0 + bnc;

  f4 rb0, rb1;
  auto asyncA = [&](int b, int k0) {
#pragma unroll
    for (int j = 0; j < 4; ++j) async_copy_b64(aoff[b][j], aptr[j] + k0);
  };
  auto gloadB = [&](int k0) {
    rb0 = *(const f4*)(bp + (size_t)k0 * D_);
    rb1 = *(const f4*)(bp + (size_t)(k0 + 1) * D_);
  };
  auto storeB = [&](int pb) {
#pragma unroll
    for (int t = 0; t < 4; ++t)
      store_bf2_cvt(&Bt[pb][bnc + t][bk2], rb0[t], rb1[t]);
  };

  v8f acc[2][2] = {};
  asyncA(0, 0);
  gloadB(0);
  storeB(0);
  WAIT_ASYNC();
  __syncthreads();
  int pb = 0;
  for (int k0 = 0; k0 < D_; k0 += BK) {
    const bool more = (k0 + BK) < D_;
    if (more) { asyncA(pb ^ 1, k0 + BK); gloadB(k0 + BK); }
    v16bf af[2], bf[2];
#pragma unroll
    for (int mi = 0; mi < 2; ++mi) af[mi] = load_frag(&As[pb][wm * 32 + mi * 16 + (lane & 15)][0], lane);
#pragma unroll
    for (int ni = 0; ni < 2; ++ni) bf[ni] = load_frag(&Bt[pb][wn * 32 + ni * 16 + (lane & 15)][0], lane);
#pragma unroll
    for (int mi = 0; mi < 2; ++mi)
#pragma unroll
      for (int ni = 0; ni < 2; ++ni)
        acc[mi][ni] = wmma_bf16(af[mi], bf[ni], acc[mi][ni]);
    if (more) storeB(pb ^ 1);
    WAIT_ASYNC();
    __syncthreads();
    pb ^= 1;
  }

#pragma unroll
  for (int mi = 0; mi < 2; ++mi)
#pragma unroll
    for (int i = 0; i < 8; ++i) {
      int lr = m0 + wm * 32 + mi * 16 + i + ((lane & 16) ? 8 : 0);
      if (lr < cnt) {
        int tok = tokens[off + lr];
        float cw = wts[off + lr];
#pragma unroll
        for (int ni = 0; ni < 2; ++ni) {
          int col = n0 + wn * 32 + ni * 16 + (lane & 15);
          atomicAdd(&out[(size_t)tok * D_ + col], cw * acc[mi][ni][i]);
        }
      }
    }
}

// =====================  Launch  =====================

extern "C" void kernel_launch(void* const* d_in, const int* in_sizes, int n_in,
                              void* d_out, int out_size, void* d_ws, size_t ws_size,
                              hipStream_t stream) {
  const float* x       = (const float*)d_in[0];
  const float* weights = (const float*)d_in[1];
  const int*   indices = (const int*)d_in[2];
  const float* Wg = (const float*)d_in[5];
  const float* Wu = (const float*)d_in[6];
  const float* Wd = (const float*)d_in[7];
  const float* sg = (const float*)d_in[8];
  const float* su = (const float*)d_in[9];
  const float* sd = (const float*)d_in[10];
  float* out = (float*)d_out;

  char* p = (char*)d_ws;
  int*   counts = (int*)p;   p += 256;
  int*   offs   = (int*)p;   p += 256;
  int*   fill   = (int*)p;   p += 256;
  int*   tokens = (int*)p;   p += CAP * sizeof(int);
  float* wts    = (float*)p; p += CAP * sizeof(float);
  size_t a = ((size_t)p + 255) & ~(size_t)255;
  p = (char*)a;
  __bf16* hs = (__bf16*)p;   p += (size_t)NT * D_ * sizeof(__bf16);   // 8 MB
  __bf16* hr = (__bf16*)p;                                            // 16 MB

  k_zero<<<1, 64, 0, stream>>>(counts, fill);
  k_count<<<CAP / 256, 256, 0, stream>>>(indices, counts);
  k_offsets<<<1, 32, 0, stream>>>(counts, offs);
  k_fill<<<CAP / 256, 256, 0, stream>>>(indices, weights, offs, fill, tokens, wts);

  dim3 gridS(D_ / BN, NT / BM);               // (16, 32)
  k_gateup_shared<<<gridS, THREADS, 0, stream>>>(x, sg, su, hs);
  k_down_shared<<<gridS, THREADS, 0, stream>>>(hs, sd, out);

  dim3 gridR(D_ / BN, CAP / BM, E_);          // (16, 64, 16)
  k_gateup_routed<<<gridR, THREADS, 0, stream>>>(x, Wg, Wu, counts, offs, tokens, hr);
  k_down_routed<<<gridR, THREADS, 0, stream>>>(hr, Wd, counts, offs, tokens, wts, out);
}
